// MemEffSelfAttn_67259187855884
// MI455X (gfx1250) — compile-verified, matrix operand order checked by reading, not run on hardware
//
#include <hip/hip_runtime.h>

// ---------------- types ----------------
typedef unsigned short ushort_t;
typedef __bf16 bf16_t;
typedef bf16_t v16bf __attribute__((ext_vector_type(16)));
typedef float  v8f   __attribute__((ext_vector_type(8)));
typedef unsigned int u32x4 __attribute__((ext_vector_type(4)));
typedef float        f32x4 __attribute__((ext_vector_type(4)));

#define T_DIM 2048
#define B_DIM 4
#define E_DIM 1024
#define H_DIM 16
#define D_DIM 64
#define F3_DIM 3072
#define R_DIM 8192   // T*B rows

// 1/sqrt(64) * log2(e): folded into Q so softmax can use exp2 directly
#define Q_SCALE 0.1803368801111204f

// fp32 -> bf16 (round-to-nearest-even)
__device__ __forceinline__ ushort_t f2bf(float f) {
  unsigned int u = __builtin_bit_cast(unsigned int, f);
  unsigned int r = (u + 0x7FFFu + ((u >> 16) & 1u)) >> 16;
  return (ushort_t)r;
}

union FragU { u32x4 q[2]; v16bf v; };

// Load a 16x32 bf16 WMMA fragment (A layout; B loads use the same pattern on
// B^T, with "row" = output column). ISA layout: lanes 0-15 hold K 0-7 (VGPR0-3)
// and K 16-23 (VGPR4-7); lanes 16-31 hold K 8-15 and K 24-31.
__device__ __forceinline__ v16bf load_frag16(const ushort_t* p0, int ld, int lane) {
  int half = lane >> 4;
  int r    = lane & 15;
  const ushort_t* p = p0 + r * ld + half * 8;
  FragU f;
  f.q[0] = *(const u32x4*)(p);        // K =  0..7  (+8 for hi half)
  f.q[1] = *(const u32x4*)(p + 16);   // K = 16..23 (+8 for hi half)
  return f.v;
}

__device__ __forceinline__ v8f wmma_bf16(v16bf a, v16bf b, v8f c) {
  return __builtin_amdgcn_wmma_f32_16x16x32_bf16(
      /*neg_a=*/false, a, /*neg_b=*/false, b,
      /*c_mod=*/(short)0, c, /*reuse_a=*/false, /*reuse_b=*/false);
}

__device__ __forceinline__ float hmax16(float v) {
  v = fmaxf(v, __shfl_xor(v, 1, 32));
  v = fmaxf(v, __shfl_xor(v, 2, 32));
  v = fmaxf(v, __shfl_xor(v, 4, 32));
  v = fmaxf(v, __shfl_xor(v, 8, 32));
  return v;
}
__device__ __forceinline__ float hsum16(float v) {
  v += __shfl_xor(v, 1, 32);
  v += __shfl_xor(v, 2, 32);
  v += __shfl_xor(v, 4, 32);
  v += __shfl_xor(v, 8, 32);
  return v;
}

// ---------------- fp32 -> bf16 conversion (8 elems/thread) ----------------
__global__ __launch_bounds__(256) void cvt_f32_bf16_kernel(
    const float* __restrict__ src, ushort_t* __restrict__ dst, int n8) {
  int i = blockIdx.x * 256 + threadIdx.x;
  if (i >= n8) return;
  const f32x4* s = (const f32x4*)src + (size_t)i * 2;
  f32x4 a = s[0], b = s[1];
  union { ushort_t h[8]; u32x4 v; } u;
  u.h[0] = f2bf(a.x); u.h[1] = f2bf(a.y); u.h[2] = f2bf(a.z); u.h[3] = f2bf(a.w);
  u.h[4] = f2bf(b.x); u.h[5] = f2bf(b.y); u.h[6] = f2bf(b.z); u.h[7] = f2bf(b.w);
  *((u32x4*)dst + i) = u.v;
}

// ---------------- QKV projection: qkv[r,f] = sum_e x[r,e]*Wqkv[f,e] ----------------
// One wave computes a 16x64 output tile; double-buffered k-chunks; results
// scattered to Q(T,D) (pre-scaled), K(T,D), V^T(D,T) per (b,h) head in bf16.
__global__ __launch_bounds__(256) void qkv_gemm_kernel(
    const ushort_t* __restrict__ xh, const ushort_t* __restrict__ wh,
    ushort_t* __restrict__ Qb, ushort_t* __restrict__ Kb, ushort_t* __restrict__ Vtb) {
  int lane = threadIdx.x & 31;
  int w    = threadIdx.x >> 5;
  int gw   = blockIdx.x * 8 + w;
  int ft   = gw % (F3_DIM / 64);
  int rt   = gw / (F3_DIM / 64);
  int r0   = rt * 16;
  int fbase = ft * 64;

  const ushort_t* pa = xh + (size_t)r0 * E_DIM;
  const ushort_t* pb = wh + (size_t)fbase * E_DIM;

  v8f acc[4] = {};
  // prime the pipeline
  v16bf a  = load_frag16(pa, E_DIM, lane);
  v16bf b0 = load_frag16(pb + (size_t)0  * E_DIM, E_DIM, lane);
  v16bf b1 = load_frag16(pb + (size_t)16 * E_DIM, E_DIM, lane);
  v16bf b2 = load_frag16(pb + (size_t)32 * E_DIM, E_DIM, lane);
  v16bf b3 = load_frag16(pb + (size_t)48 * E_DIM, E_DIM, lane);

  for (int e0 = 0; e0 < E_DIM - 32; e0 += 32) {
    int e1 = e0 + 32;
    v16bf an  = load_frag16(pa + e1, E_DIM, lane);
    v16bf bn0 = load_frag16(pb + (size_t)0  * E_DIM + e1, E_DIM, lane);
    v16bf bn1 = load_frag16(pb + (size_t)16 * E_DIM + e1, E_DIM, lane);
    v16bf bn2 = load_frag16(pb + (size_t)32 * E_DIM + e1, E_DIM, lane);
    v16bf bn3 = load_frag16(pb + (size_t)48 * E_DIM + e1, E_DIM, lane);
    acc[0] = wmma_bf16(a, b0, acc[0]);
    acc[1] = wmma_bf16(a, b1, acc[1]);
    acc[2] = wmma_bf16(a, b2, acc[2]);
    acc[3] = wmma_bf16(a, b3, acc[3]);
    a = an; b0 = bn0; b1 = bn1; b2 = bn2; b3 = bn3;
  }
  acc[0] = wmma_bf16(a, b0, acc[0]);
  acc[1] = wmma_bf16(a, b1, acc[1]);
  acc[2] = wmma_bf16(a, b2, acc[2]);
  acc[3] = wmma_bf16(a, b3, acc[3]);

  int half = lane >> 4, col = lane & 15;
  int which = fbase >> 10;           // 64-wide tile never straddles Q/K/V
  int h     = (fbase & 1023) >> 6;   // uniform per wave (tile inside one head)
  int bh0   = h;                     // + b*H below

  if (which == 0) {                  // Q, pre-scaled by 1/sqrt(d)*log2(e)
#pragma unroll
    for (int g = 0; g < 8; g++) {
      int r = r0 + g + 8 * half;
      int t = r >> 2, b = r & 3;
      size_t base = ((size_t)(b * H_DIM + bh0) * T_DIM + t) * D_DIM;
#pragma unroll
      for (int nt = 0; nt < 4; nt++)
        Qb[base + nt * 16 + col] = f2bf(acc[nt][g] * Q_SCALE);
    }
  } else if (which == 1) {           // K
#pragma unroll
    for (int g = 0; g < 8; g++) {
      int r = r0 + g + 8 * half;
      int t = r >> 2, b = r & 3;
      size_t base = ((size_t)(b * H_DIM + bh0) * T_DIM + t) * D_DIM;
#pragma unroll
      for (int nt = 0; nt < 4; nt++)
        Kb[base + nt * 16 + col] = f2bf(acc[nt][g]);
    }
  } else {                           // V stored transposed: (D,T) per head
#pragma unroll
    for (int g = 0; g < 8; g++) {
      int r = r0 + g + 8 * half;
      int t = r >> 2, b = r & 3;
      size_t base = (size_t)(b * H_DIM + bh0) * D_DIM * T_DIM + t;
#pragma unroll
      for (int nt = 0; nt < 4; nt++)
        Vtb[base + (size_t)(nt * 16 + col) * T_DIM] = f2bf(acc[nt][g]);
    }
  }
}

// ---------------- flash attention: one wave per 16-query tile ----------------
__global__ __launch_bounds__(256) void attn_kernel(
    const ushort_t* __restrict__ Q, const ushort_t* __restrict__ K,
    const ushort_t* __restrict__ Vt, const int* __restrict__ mask,
    ushort_t* __restrict__ ctx) {
  __shared__ __align__(16) ushort_t ldsP[8 * 16 * 32];
  int lane = threadIdx.x & 31;
  int w    = threadIdx.x >> 5;
  int gw   = blockIdx.x * 8 + w;
  int qt = gw & 127;             // 128 q-tiles per (b,h)
  int bh = gw >> 7;
  int b  = bh >> 4;
  int h  = bh & 15;
  int half = lane >> 4, col = lane & 15;

  const ushort_t* Qbase = Q + (size_t)bh * T_DIM * D_DIM + (size_t)qt * 16 * D_DIM;
  v16bf qf0 = load_frag16(Qbase,      D_DIM, lane);   // head-dim 0..31 (pre-scaled)
  v16bf qf1 = load_frag16(Qbase + 32, D_DIM, lane);   // head-dim 32..63
  const ushort_t* Kbase = K  + (size_t)bh * T_DIM * D_DIM;
  const ushort_t* Vbase = Vt + (size_t)bh * D_DIM * T_DIM;
  const int* kmask = mask + b * T_DIM;
  ushort_t* myP = ldsP + w * (16 * 32);

  v8f o0 = {}, o1 = {}, o2 = {}, o3 = {};
  float mrow[8], lrow[8];
#pragma unroll
  for (int g = 0; g < 8; g++) { mrow[g] = -1e30f; lrow[g] = 0.f; }

  // prime K fragments for kt = 0
  v16bf kf0 = load_frag16(Kbase,                   D_DIM, lane);
  v16bf kf1 = load_frag16(Kbase + 32,              D_DIM, lane);
  v16bf kf2 = load_frag16(Kbase + 16 * D_DIM,      D_DIM, lane);
  v16bf kf3 = load_frag16(Kbase + 16 * D_DIM + 32, D_DIM, lane);

  for (int kt = 0; kt < T_DIM; kt += 32) {
    // ---- V fragments: independent of P, issue early to overlap softmax ----
    const ushort_t* Vtile = Vbase + kt;
    v16bf vf0 = load_frag16(Vtile,              T_DIM, lane);
    v16bf vf1 = load_frag16(Vtile + 16 * T_DIM, T_DIM, lane);
    v16bf vf2 = load_frag16(Vtile + 32 * T_DIM, T_DIM, lane);
    v16bf vf3 = load_frag16(Vtile + 48 * T_DIM, T_DIM, lane);

    // ---- S = Q * K^T for 32 keys (two 16x16 C tiles) ----
    v8f s0 = {}, s1 = {};
    s0 = wmma_bf16(qf0, kf0, s0);
    s0 = wmma_bf16(qf1, kf1, s0);
    s1 = wmma_bf16(qf0, kf2, s1);
    s1 = wmma_bf16(qf1, kf3, s1);

    // ---- prefetch next iteration's K fragments (wrap harmlessly at end) ----
    int ktn = (kt + 32) & (T_DIM - 1);
    const ushort_t* Ktn = Kbase + (size_t)ktn * D_DIM;
    kf0 = load_frag16(Ktn,                   D_DIM, lane);
    kf1 = load_frag16(Ktn + 32,              D_DIM, lane);
    kf2 = load_frag16(Ktn + 16 * D_DIM,      D_DIM, lane);
    kf3 = load_frag16(Ktn + 16 * D_DIM + 32, D_DIM, lane);

    bool pad0 = kmask[kt + col] != 0;
    bool pad1 = kmask[kt + 16 + col] != 0;

    // ---- online softmax in base-2 domain (scale folded into Q) ----
#pragma unroll
    for (int g = 0; g < 8; g++) {
      float a0 = pad0 ? -1e30f : s0[g];
      float a1 = pad1 ? -1e30f : s1[g];
      float tmax = hmax16(fmaxf(a0, a1));
      float mn   = fmaxf(mrow[g], tmax);
      float p0 = exp2f(a0 - mn);
      float p1 = exp2f(a1 - mn);
      float alpha = exp2f(mrow[g] - mn);
      lrow[g] = lrow[g] * alpha + hsum16(p0 + p1);
      mrow[g] = mn;
      o0[g] *= alpha; o1[g] *= alpha; o2[g] *= alpha; o3[g] *= alpha;
      int rowoff = (g + 8 * half) * 32;
      myP[rowoff + col]      = f2bf(p0);           // C-layout -> row-major P in LDS
      myP[rowoff + 16 + col] = f2bf(p1);
    }
    asm volatile("s_wait_dscnt 0" ::: "memory");   // P stores visible before reload

    // ---- O += P * V (A-frag reload from LDS, B-frags prefetched above) ----
    v16bf pf = load_frag16(myP, 32, lane);
    o0 = wmma_bf16(pf, vf0, o0);
    o1 = wmma_bf16(pf, vf1, o1);
    o2 = wmma_bf16(pf, vf2, o2);
    o3 = wmma_bf16(pf, vf3, o3);
  }

  // ---- normalize and store ctx[b][t][h*64+d] as bf16 ----
#pragma unroll
  for (int g = 0; g < 8; g++) {
    int m = g + 8 * half;
    int t = qt * 16 + m;
    float inv = 1.0f / lrow[g];
    size_t base = ((size_t)b * T_DIM + t) * E_DIM + h * D_DIM + col;
    ctx[base +  0] = f2bf(o0[g] * inv);
    ctx[base + 16] = f2bf(o1[g] * inv);
    ctx[base + 32] = f2bf(o2[g] * inv);
    ctx[base + 48] = f2bf(o3[g] * inv);
  }
}

// ---------------- out proj: out[t,b,f] = sum_e ctx[b,t,e]*Wo[f,e] ----------------
__global__ __launch_bounds__(256) void oproj_kernel(
    const ushort_t* __restrict__ ctx, const ushort_t* __restrict__ woh,
    float* __restrict__ out) {
  int lane = threadIdx.x & 31;
  int w    = threadIdx.x >> 5;
  int gw   = blockIdx.x * 8 + w;
  int ft = gw & 15;              // 16 column tiles of 64
  int rt = gw >> 4;
  int r0 = rt * 16;
  int fbase = ft * 64;

  const ushort_t* pa = ctx + (size_t)r0 * E_DIM;
  const ushort_t* pb = woh + (size_t)fbase * E_DIM;

  v8f acc[4] = {};
  v16bf a  = load_frag16(pa, E_DIM, lane);
  v16bf b0 = load_frag16(pb + (size_t)0  * E_DIM, E_DIM, lane);
  v16bf b1 = load_frag16(pb + (size_t)16 * E_DIM, E_DIM, lane);
  v16bf b2 = load_frag16(pb + (size_t)32 * E_DIM, E_DIM, lane);
  v16bf b3 = load_frag16(pb + (size_t)48 * E_DIM, E_DIM, lane);

  for (int e0 = 0; e0 < E_DIM - 32; e0 += 32) {
    int e1 = e0 + 32;
    v16bf an  = load_frag16(pa + e1, E_DIM, lane);
    v16bf bn0 = load_frag16(pb + (size_t)0  * E_DIM + e1, E_DIM, lane);
    v16bf bn1 = load_frag16(pb + (size_t)16 * E_DIM + e1, E_DIM, lane);
    v16bf bn2 = load_frag16(pb + (size_t)32 * E_DIM + e1, E_DIM, lane);
    v16bf bn3 = load_frag16(pb + (size_t)48 * E_DIM + e1, E_DIM, lane);
    acc[0] = wmma_bf16(a, b0, acc[0]);
    acc[1] = wmma_bf16(a, b1, acc[1]);
    acc[2] = wmma_bf16(a, b2, acc[2]);
    acc[3] = wmma_bf16(a, b3, acc[3]);
    a = an; b0 = bn0; b1 = bn1; b2 = bn2; b3 = bn3;
  }
  acc[0] = wmma_bf16(a, b0, acc[0]);
  acc[1] = wmma_bf16(a, b1, acc[1]);
  acc[2] = wmma_bf16(a, b2, acc[2]);
  acc[3] = wmma_bf16(a, b3, acc[3]);

  int half = lane >> 4, col = lane & 15;
#pragma unroll
  for (int g = 0; g < 8; g++) {
    int r = r0 + g + 8 * half;     // r = b*T + t
    int b = r >> 11;
    int t = r & 2047;
#pragma unroll
    for (int nt = 0; nt < 4; nt++) {
      out[((size_t)t * B_DIM + b) * E_DIM + fbase + nt * 16 + col] = acc[nt][g];
    }
  }
}

// ---------------- host launcher ----------------
extern "C" void kernel_launch(void* const* d_in, const int* in_sizes, int n_in,
                              void* d_out, int out_size, void* d_ws, size_t ws_size,
                              hipStream_t stream) {
  (void)in_sizes; (void)n_in; (void)out_size; (void)ws_size;
  const float* x    = (const float*)d_in[0];   // (T,B,E)
  const float* Wqkv = (const float*)d_in[1];   // (3E,E)
  const float* Wo   = (const float*)d_in[2];   // (E,E)
  const int*   mask = (const int*)d_in[3];     // (B,T), nonzero = pad
  float* out = (float*)d_out;                  // (T,B,E)

  // workspace carve-up (bf16 buffers, element offsets)
  ushort_t* xh    = (ushort_t*)d_ws;                       // R*E
  ushort_t* wqkvh = xh    + (size_t)R_DIM * E_DIM;         // 3E*E
  ushort_t* woh   = wqkvh + (size_t)F3_DIM * E_DIM;        // E*E
  ushort_t* Qb    = woh   + (size_t)E_DIM * E_DIM;         // B*H*T*D
  ushort_t* Kb    = Qb    + (size_t)B_DIM * H_DIM * T_DIM * D_DIM;
  ushort_t* Vtb   = Kb    + (size_t)B_DIM * H_DIM * T_DIM * D_DIM;
  ushort_t* ctxh  = Vtb   + (size_t)B_DIM * H_DIM * T_DIM * D_DIM;  // R*E

  // 1) fp32 -> bf16 conversions
  {
    int n8 = (R_DIM * E_DIM) / 8;
    cvt_f32_bf16_kernel<<<(n8 + 255) / 256, 256, 0, stream>>>(x, xh, n8);
  }
  {
    int n8 = (F3_DIM * E_DIM) / 8;
    cvt_f32_bf16_kernel<<<(n8 + 255) / 256, 256, 0, stream>>>(Wqkv, wqkvh, n8);
  }
  {
    int n8 = (E_DIM * E_DIM) / 8;
    cvt_f32_bf16_kernel<<<(n8 + 255) / 256, 256, 0, stream>>>(Wo, woh, n8);
  }

  // 2) QKV projection: (R/16)*(3E/64) = 512*48 wave-tiles, 8 waves/block
  qkv_gemm_kernel<<<(512 * 48) / 8, 256, 0, stream>>>(xh, wqkvh, Qb, Kb, Vtb);

  // 3) attention: B*H*(T/16) = 8192 wave-tiles, 8 waves/block
  attn_kernel<<<8192 / 8, 256, 0, stream>>>(Qb, Kb, Vtb, mask, ctxh);

  // 4) output projection: (R/16)*(E/64) = 512*16 wave-tiles
  oproj_kernel<<<(512 * 16) / 8, 256, 0, stream>>>(ctxh, woh, out);
}
